// EncoderBlock_11003706213312
// MI455X (gfx1250) — matrix operand order repaired and checked by reading, not compiled, not run on hardware
//
#include <hip/hip_runtime.h>
#include <math.h>
#include <stdint.h>

typedef __attribute__((ext_vector_type(16))) _Float16 v16h;
typedef __attribute__((ext_vector_type(8)))  float    v8f;
typedef __attribute__((ext_vector_type(4)))  int      v4i;
typedef _Float16 half_t;

#define D_MODEL 1024
#define NHEAD   16
#define DK      64
#define D_FF    4096
#define SEQ     2048
#define BATCH   4
#define NTOK    (BATCH * SEQ)

// gfx1250 async global->LDS path, guarded so compile is safe either way
#if defined(__has_builtin)
#if __has_builtin(__builtin_amdgcn_global_load_async_to_lds_b128) && \
    __has_builtin(__builtin_amdgcn_s_wait_asynccnt)
#define USE_ASYNC_LDS 1
#endif
#endif
#ifndef USE_ASYNC_LDS
#define USE_ASYNC_LDS 0
#endif

#if USE_ASYNC_LDS
typedef __attribute__((address_space(1))) v4i as1_v4i;
typedef __attribute__((address_space(3))) v4i as3_v4i;
// async 16B copy: global -> LDS, tracked by ASYNCcnt
__device__ inline void async_copy_b128(const half_t* gsrc, half_t* ldst) {
    __builtin_amdgcn_global_load_async_to_lds_b128(
        (as1_v4i*)(uintptr_t)gsrc, (as3_v4i*)(uintptr_t)ldst, 0, 0);
}
#endif

// ---------------------------------------------------------------------------
// WMMA fragment loaders (layouts per CDNA5 ISA 7.12.2, wave32)
// ---------------------------------------------------------------------------

// A fragment: 16x32 f16, row-major source, base = &A[row0*lda + k0]
__device__ inline v16h frag_a(const half_t* __restrict__ base, int lda, int lane) {
    const int m  = lane & 15;
    const int kb = (lane < 16) ? 0 : 8;
    const half_t* p = base + (size_t)m * lda + kb;
    v16h a;
#pragma unroll
    for (int v = 0; v < 4; ++v) {           // VGPR 0..3 : K = kb + 2v + {0,1}
        a[2 * v]     = p[2 * v];
        a[2 * v + 1] = p[2 * v + 1];
    }
#pragma unroll
    for (int v = 0; v < 4; ++v) {           // VGPR 4..7 : K = 16 + kb + 2v + {0,1}
        a[8 + 2 * v] = p[16 + 2 * v];
        a[9 + 2 * v] = p[16 + 2 * v + 1];
    }
    return a;
}

// B fragment from transposed storage: logical B[k][n] = Bt[n*ldb + k],
// base = &Bt[col0*ldb + k0]  (contiguous in k -> coalesced 32B per lane)
__device__ inline v16h frag_bt(const half_t* __restrict__ base, int ldb, int lane) {
    const int n  = lane & 15;
    const int kb = (lane < 16) ? 0 : 8;
    const half_t* p = base + (size_t)n * ldb + kb;
    v16h b;
#pragma unroll
    for (int e = 0; e < 16; ++e) b[e] = p[(e < 8) ? e : e + 8];
    return b;
}

__device__ inline v8f wmma_f16(v16h a, v16h b, v8f c) {
    return __builtin_amdgcn_wmma_f32_16x16x32_f16(false, a, false, b, (short)0, c,
                                                  false, false);
}

// ---------------------------------------------------------------------------
// fp32 [K x N] -> fp16 [N x K] transpose-cast (LDS tiled, coalesced both sides)
// ---------------------------------------------------------------------------
__global__ __launch_bounds__(256) void cast_transpose(
    const float* __restrict__ in, half_t* __restrict__ out, int K, int N)
{
    __shared__ float tile[32][33];
    const int k0 = blockIdx.y * 32;
    const int n0 = blockIdx.x * 32;
    const int tx = threadIdx.x & 31;
    const int ty = threadIdx.x >> 5;           // 0..7
#pragma unroll
    for (int i = 0; i < 32; i += 8)
        tile[ty + i][tx] = in[(size_t)(k0 + ty + i) * N + n0 + tx];
    __syncthreads();
#pragma unroll
    for (int i = 0; i < 32; i += 8)
        out[(size_t)(n0 + ty + i) * K + k0 + tx] = (half_t)tile[tx][ty + i];
}

// ---------------------------------------------------------------------------
// LayerNorm (torch-style: unbiased std, eps added to std), fp16 output
// ---------------------------------------------------------------------------
__global__ __launch_bounds__(256) void layernorm_f16(
    const float* __restrict__ x, half_t* __restrict__ out,
    const float* __restrict__ alpha, const float* __restrict__ beta)
{
    __shared__ float red1[256];
    __shared__ float red2[256];
    const int tid = threadIdx.x;
    const size_t row = blockIdx.x;
    const float4 xv = ((const float4*)(x + row * D_MODEL))[tid];

    red1[tid] = xv.x + xv.y + xv.z + xv.w;
    red2[tid] = xv.x * xv.x + xv.y * xv.y + xv.z * xv.z + xv.w * xv.w;
    __syncthreads();
#pragma unroll
    for (int off = 128; off > 0; off >>= 1) {
        if (tid < off) { red1[tid] += red1[tid + off]; red2[tid] += red2[tid + off]; }
        __syncthreads();
    }
    const float mean = red1[0] * (1.0f / D_MODEL);
    const float var  = fmaxf(red2[0] - red1[0] * red1[0] * (1.0f / D_MODEL), 0.0f)
                       * (1.0f / (D_MODEL - 1));
    const float inv  = alpha[0] / (sqrtf(var) + 1e-6f);
    const float bb   = beta[0];

    half_t* o = out + row * D_MODEL + tid * 4;
    o[0] = (half_t)((xv.x - mean) * inv + bb);
    o[1] = (half_t)((xv.y - mean) * inv + bb);
    o[2] = (half_t)((xv.z - mean) * inv + bb);
    o[3] = (half_t)((xv.w - mean) * inv + bb);
}

// ---------------------------------------------------------------------------
// WMMA GEMM: C[M,N] = epi( A[M,K] @ Bt[N,K]^T )
// 8 waves/block, block tile 128(M) x 64(N), wave tile 32x32 (2x2 frags),
// K step 32 -> 4 WMMAs per step on 128B of fragments per lane.
// ---------------------------------------------------------------------------
__global__ __launch_bounds__(256) void gemm_wmma(
    const half_t* __restrict__ A, int lda,
    const half_t* __restrict__ Bt, int ldb,       // ldb == K
    const float* __restrict__ bias,
    const float* __restrict__ residual,
    float* __restrict__ outf, half_t* __restrict__ outh,
    int M, int N, int K, float scale, int relu)
{
    const int lane = threadIdx.x & 31;
    const int wave = threadIdx.x >> 5;
    const int row0 = blockIdx.y * 128 + (wave & 3) * 32;
    const int col0 = blockIdx.x * 64  + (wave >> 2) * 32;

    v8f acc00 = {}, acc01 = {}, acc10 = {}, acc11 = {};
    const half_t* A0 = A  + (size_t)row0 * lda;
    const half_t* A1 = A  + (size_t)(row0 + 16) * lda;
    const half_t* B0 = Bt + (size_t)col0 * ldb;
    const half_t* B1 = Bt + (size_t)(col0 + 16) * ldb;

    for (int k0 = 0; k0 < K; k0 += 32) {
        __builtin_prefetch(A0 + k0 + 128, 0, 0);   // -> global_prefetch_b8
        v16h a0 = frag_a (A0 + k0, lda, lane);
        v16h a1 = frag_a (A1 + k0, lda, lane);
        v16h b0 = frag_bt(B0 + k0, ldb, lane);
        v16h b1 = frag_bt(B1 + k0, ldb, lane);
        acc00 = wmma_f16(a0, b0, acc00);
        acc01 = wmma_f16(a0, b1, acc01);
        acc10 = wmma_f16(a1, b0, acc10);
        acc11 = wmma_f16(a1, b1, acc11);
    }

    const int hs = lane >> 4;
    const int nn = lane & 15;
    const float bn0 = bias ? bias[col0 + nn]      : 0.0f;
    const float bn1 = bias ? bias[col0 + 16 + nn] : 0.0f;
#pragma unroll
    for (int r = 0; r < 8; ++r) {
        const int m0 = row0 + r + 8 * hs;
        const int m1 = m0 + 16;
        float v00 = (acc00[r] + bn0) * scale;
        float v01 = (acc01[r] + bn1) * scale;
        float v10 = (acc10[r] + bn0) * scale;
        float v11 = (acc11[r] + bn1) * scale;
        if (relu) {
            v00 = fmaxf(v00, 0.0f); v01 = fmaxf(v01, 0.0f);
            v10 = fmaxf(v10, 0.0f); v11 = fmaxf(v11, 0.0f);
        }
        const size_t o00 = (size_t)m0 * N + col0 + nn;
        const size_t o01 = o00 + 16;
        const size_t o10 = (size_t)m1 * N + col0 + nn;
        const size_t o11 = o10 + 16;
        if (residual) {
            v00 += residual[o00]; v01 += residual[o01];
            v10 += residual[o10]; v11 += residual[o11];
        }
        if (outf) {
            outf[o00] = v00; outf[o01] = v01; outf[o10] = v10; outf[o11] = v11;
        }
        if (outh) {
            outh[o00] = (half_t)v00; outh[o01] = (half_t)v01;
            outh[o10] = (half_t)v10; outh[o11] = (half_t)v11;
        }
    }
}

// ---------------------------------------------------------------------------
// Flash attention. One wave = 16 query rows; block = 128 query rows, one (b,h).
// K tile (32 keys x 64) staged in LDS (async global->LDS when available),
// V tile staged transposed in LDS. P tile bounced through per-wave LDS to
// become an A fragment. 1/sqrt(dk) pre-folded into Q.
// ---------------------------------------------------------------------------
__global__ __launch_bounds__(256) void flash_attn(
    const half_t* __restrict__ Q, const half_t* __restrict__ K,
    const half_t* __restrict__ V, half_t* __restrict__ O)
{
    __shared__ half_t ldsK[32 * 64];        // [key][d]
    __shared__ half_t ldsVt[64 * 32];       // [d][key]
    __shared__ half_t lds_p[8][16 * 32];    // per-wave P tile

    const int lane = threadIdx.x & 31;
    const int wave = threadIdx.x >> 5;
    const int h    = blockIdx.y;
    const int b    = blockIdx.z;
    const int q0   = blockIdx.x * 128 + wave * 16;

    const size_t base = ((size_t)b * SEQ) * D_MODEL + (size_t)h * DK;
    const half_t* Qb = Q + base;
    const half_t* Kb = K + base;
    const half_t* Vb = V + base;

    const v16h qf0 = frag_a(Qb + (size_t)q0 * D_MODEL + 0,  D_MODEL, lane);
    const v16h qf1 = frag_a(Qb + (size_t)q0 * D_MODEL + 32, D_MODEL, lane);

    v8f o0 = {}, o1 = {}, o2 = {}, o3 = {};
    float mi[8], li[8];
#pragma unroll
    for (int r = 0; r < 8; ++r) { mi[r] = -3.0e38f; li[r] = 0.0f; }

    half_t* lp = &lds_p[wave][0];
    const int nn = lane & 15;
    const int hs = lane >> 4;

    // cooperative-copy coordinates: 256 threads x 8 halves = one 32x64 tile
    const int cidx = threadIdx.x * 8;
    const int ckey = cidx >> 6;
    const int ccol = cidx & 63;

    for (int kb = 0; kb < SEQ; kb += 32) {
        __syncthreads();   // previous iteration's LDS consumers are done
        {
            const half_t* ksrc = Kb + (size_t)(kb + ckey) * D_MODEL + ccol;
#if USE_ASYNC_LDS
            async_copy_b128(ksrc, &ldsK[cidx]);
#else
            *(uint4*)&ldsK[cidx] = *(const uint4*)ksrc;
#endif
            const half_t* vsrc = Vb + (size_t)(kb + ckey) * D_MODEL + ccol;
#pragma unroll
            for (int j = 0; j < 8; ++j)              // transpose into [d][key]
                ldsVt[(ccol + j) * 32 + ckey] = vsrc[j];
#if USE_ASYNC_LDS
            __builtin_amdgcn_s_wait_asynccnt(0);
#endif
        }
        __syncthreads();

        // ---- scores: two 16-key tiles, contraction over d_k = 64 (2 steps)
        v8f s0 = {}, s1 = {};
        s0 = wmma_f16(qf0, frag_bt(ldsK + 0 * 64 + 0,   64, lane), s0);
        s0 = wmma_f16(qf1, frag_bt(ldsK + 0 * 64 + 32,  64, lane), s0);
        s1 = wmma_f16(qf0, frag_bt(ldsK + 16 * 64 + 0,  64, lane), s1);
        s1 = wmma_f16(qf1, frag_bt(ldsK + 16 * 64 + 32, 64, lane), s1);

        // ---- online softmax (row stats replicated across each 16-lane half)
#pragma unroll
        for (int r = 0; r < 8; ++r) {
            float rm = fmaxf(s0[r], s1[r]);
#pragma unroll
            for (int msk = 8; msk >= 1; msk >>= 1) rm = fmaxf(rm, __shfl_xor(rm, msk));
            const float mnew = fmaxf(mi[r], rm);
            const float corr = __expf(mi[r] - mnew);
            const float p0 = __expf(s0[r] - mnew);
            const float p1 = __expf(s1[r] - mnew);
            float rs = p0 + p1;
#pragma unroll
            for (int msk = 8; msk >= 1; msk >>= 1) rs += __shfl_xor(rs, msk);
            li[r] = li[r] * corr + rs;
            mi[r] = mnew;
            o0[r] *= corr; o1[r] *= corr; o2[r] *= corr; o3[r] *= corr;
            s0[r] = p0; s1[r] = p1;
        }

        // ---- C-layout -> A-fragment layout via per-wave LDS bounce
#pragma unroll
        for (int r = 0; r < 8; ++r) {
            const int m = r + 8 * hs;
            lp[m * 32 + nn]      = (half_t)s0[r];
            lp[m * 32 + 16 + nn] = (half_t)s1[r];
        }
        const v16h pa = frag_a(lp, 32, lane);

        // ---- accumulate P @ V (contraction over 32 keys, V^T in LDS)
        o0 = wmma_f16(pa, frag_bt(ldsVt + 0  * 32, 32, lane), o0);
        o1 = wmma_f16(pa, frag_bt(ldsVt + 16 * 32, 32, lane), o1);
        o2 = wmma_f16(pa, frag_bt(ldsVt + 32 * 32, 32, lane), o2);
        o3 = wmma_f16(pa, frag_bt(ldsVt + 48 * 32, 32, lane), o3);
    }

    // ---- normalize and write context (fp16, feeds O-projection GEMM)
#pragma unroll
    for (int r = 0; r < 8; ++r) {
        const float inv = 1.0f / li[r];
        const int t = q0 + r + 8 * hs;
        half_t* o = O + ((size_t)b * SEQ + t) * D_MODEL + (size_t)h * DK;
        o[0  + nn] = (half_t)(o0[r] * inv);
        o[16 + nn] = (half_t)(o1[r] * inv);
        o[32 + nn] = (half_t)(o2[r] * inv);
        o[48 + nn] = (half_t)(o3[r] * inv);
    }
}

// ---------------------------------------------------------------------------
// Host-side orchestration (graph-capture safe: launches only)
// ---------------------------------------------------------------------------
extern "C" void kernel_launch(void* const* d_in, const int* in_sizes, int n_in,
                              void* d_out, int out_size, void* d_ws, size_t ws_size,
                              hipStream_t stream)
{
    (void)in_sizes; (void)n_in; (void)out_size; (void)ws_size;
    const float* x      = (const float*)d_in[0];
    const float* wq     = (const float*)d_in[1];
    const float* bq     = (const float*)d_in[2];
    const float* wk     = (const float*)d_in[3];
    const float* bk     = (const float*)d_in[4];
    const float* wv     = (const float*)d_in[5];
    const float* bvv    = (const float*)d_in[6];
    const float* wo     = (const float*)d_in[7];
    const float* bo     = (const float*)d_in[8];
    const float* w1     = (const float*)d_in[9];
    const float* b1     = (const float*)d_in[10];
    const float* w2     = (const float*)d_in[11];
    const float* b2     = (const float*)d_in[12];
    const float* alpha1 = (const float*)d_in[13];
    const float* bias1  = (const float*)d_in[14];
    const float* alpha2 = (const float*)d_in[15];
    const float* bias2  = (const float*)d_in[16];
    float* out = (float*)d_out;

    char* ws = (char*)d_ws;
    size_t off = 0;
    auto bump = [&](size_t bytes) -> void* {
        void* p = ws + off;
        off += (bytes + 255) & ~(size_t)255;
        return p;
    };
    half_t* wqt  = (half_t*)bump((size_t)D_MODEL * D_MODEL * 2);
    half_t* wkt  = (half_t*)bump((size_t)D_MODEL * D_MODEL * 2);
    half_t* wvt  = (half_t*)bump((size_t)D_MODEL * D_MODEL * 2);
    half_t* wot  = (half_t*)bump((size_t)D_MODEL * D_MODEL * 2);
    half_t* w1t  = (half_t*)bump((size_t)D_MODEL * D_FF * 2);
    half_t* w2t  = (half_t*)bump((size_t)D_FF * D_MODEL * 2);
    half_t* ln1  = (half_t*)bump((size_t)NTOK * D_MODEL * 2);
    half_t* q16  = (half_t*)bump((size_t)NTOK * D_MODEL * 2);
    half_t* k16  = (half_t*)bump((size_t)NTOK * D_MODEL * 2);
    half_t* v16  = (half_t*)bump((size_t)NTOK * D_MODEL * 2);
    half_t* ctx  = (half_t*)bump((size_t)NTOK * D_MODEL * 2);
    float*  hbuf = (float*)bump((size_t)NTOK * D_MODEL * 4);
    half_t* ln2  = (half_t*)bump((size_t)NTOK * D_MODEL * 2);
    half_t* ffn1 = (half_t*)bump((size_t)NTOK * D_FF * 2);

    auto tcast = [&](const float* src, half_t* dst, int K, int N) {
        cast_transpose<<<dim3(N / 32, K / 32), 256, 0, stream>>>(src, dst, K, N);
    };
    auto gemm = [&](const half_t* A, int lda, const half_t* Bt,
                    const float* bias, const float* res, float* of, half_t* oh,
                    int M, int N, int K, float scale, int relu) {
        dim3 grid(N / 64, M / 128);
        gemm_wmma<<<grid, 256, 0, stream>>>(A, lda, Bt, K, bias, res, of, oh,
                                            M, N, K, scale, relu);
    };

    // weights -> fp16, transposed to [N x K]
    tcast(wq, wqt, D_MODEL, D_MODEL);
    tcast(wk, wkt, D_MODEL, D_MODEL);
    tcast(wv, wvt, D_MODEL, D_MODEL);
    tcast(wo, wot, D_MODEL, D_MODEL);
    tcast(w1, w1t, D_MODEL, D_FF);
    tcast(w2, w2t, D_FF, D_MODEL);

    // LN1
    layernorm_f16<<<NTOK, 256, 0, stream>>>(x, ln1, alpha1, bias1);

    // Q/K/V projections (1/sqrt(64) folded into Q)
    gemm(ln1, D_MODEL, wqt, bq,  nullptr, nullptr, q16, NTOK, D_MODEL, D_MODEL, 0.125f, 0);
    gemm(ln1, D_MODEL, wkt, bk,  nullptr, nullptr, k16, NTOK, D_MODEL, D_MODEL, 1.0f,   0);
    gemm(ln1, D_MODEL, wvt, bvv, nullptr, nullptr, v16, NTOK, D_MODEL, D_MODEL, 1.0f,   0);

    // fused attention
    flash_attn<<<dim3(SEQ / 128, NHEAD, BATCH), 256, 0, stream>>>(q16, k16, v16, ctx);

    // O-projection + residual -> h (fp32)
    gemm(ctx, D_MODEL, wot, bo, x, hbuf, nullptr, NTOK, D_MODEL, D_MODEL, 1.0f, 0);

    // LN2
    layernorm_f16<<<NTOK, 256, 0, stream>>>(hbuf, ln2, alpha2, bias2);

    // FFN
    gemm(ln2,  D_MODEL, w1t, b1, nullptr, nullptr, ffn1, NTOK, D_FF,    D_MODEL, 1.0f, 1);
    gemm(ffn1, D_FF,    w2t, b2, hbuf,    out,     nullptr, NTOK, D_MODEL, D_FF,  1.0f, 0);
}
